// CustomGraphConv_24988119728418
// MI455X (gfx1250) — matrix operand order repaired and checked by reading, not compiled, not run on hardware
//
#include <hip/hip_runtime.h>

// ===========================================================================
// CustomGraphConv for MI455X (gfx1250, wave32, WMMA).
//
// Roofline: ~82 GFLOP vs ~0.5-1 GB HBM traffic (~40us @ 23.3 TB/s) ->
// balanced; matrix math must run on v_wmma. fp32 accuracy is recovered from
// f16 WMMA via operand splitting (x = hi + f16(x-hi)):
//    A@B ~= Ah@Bh + Ah@Bl + Al@Bh   (error ~2^-22 relative, f32 accumulate)
//
// v2: LDS operands are stored as two *deinterleaved* f16 planes (hi / lo),
// so every B fragment is two 16B ds reads (v8h) straight into a v16h
// (zero unpack VALU), and layer-2 A fragments are two 16B v8h reads. This
// removes the v_cvt/v_mov unpack storm that made v1 VALU-bound
// (valu=2842 vs wmma=96 in the round-1 histogram).
//
// d_out doubles as the scatter accumulator: init copies node_feats into it
// (residual baked in), edge kernel atomically adds messages, node kernel
// rewrites it in place. d_ws unused -> trivially deterministic on replay.
// ===========================================================================

typedef __attribute__((ext_vector_type(16))) _Float16 v16h;
typedef __attribute__((ext_vector_type(8)))  _Float16 v8h;
typedef __attribute__((ext_vector_type(8)))  float    v8f;

namespace {
constexpr int kInFeat = 128;
// Strides in halfs. Rows stay 32B-aligned; row stride mod 64 dwords == 8,
// spreading the 16-lane x 32B fragment reads over all LDS banks.
constexpr int kStA = 272;  // K=256 rows (w1a planes)
constexpr int kStB = 144;  // K=128 rows (w1b/w2a/w2b planes, t staging)

// Edge-kernel LDS layout (offsets in halfs, all 32B-aligned):
constexpr int kE_W1aHi = 0;
constexpr int kE_W1aLo = kE_W1aHi + 128 * kStA;
constexpr int kE_W1bHi = kE_W1aLo + 128 * kStA;
constexpr int kE_W1bLo = kE_W1bHi + 128 * kStB;
constexpr int kE_THi   = kE_W1bLo + 128 * kStB;   // 8 waves x 16 x kStB
constexpr int kE_TLo   = kE_THi + 8 * 16 * kStB;
constexpr int kEdgeLds = (kE_TLo + 8 * 16 * kStB) * 2;  // 286,720 B < 320 KB

// Node-kernel LDS layout:
constexpr int kN_W2aHi = 0;
constexpr int kN_W2aLo = kN_W2aHi + 128 * kStB;
constexpr int kN_W2bHi = kN_W2aLo + 128 * kStB;
constexpr int kN_W2bLo = kN_W2bHi + 128 * kStB;
constexpr int kN_THi   = kN_W2bLo + 128 * kStB;
constexpr int kN_TLo   = kN_THi + 8 * 16 * kStB;
constexpr int kNodeLds = (kN_TLo + 8 * 16 * kStB) * 2;  // 221,184 B
}

// Three f16 WMMAs emulating one near-fp32 WMMA (split operands).
__device__ __forceinline__ v8f wmma3(v16h ah, v16h al, v16h bh, v16h bl, v8f c) {
  c = __builtin_amdgcn_wmma_f32_16x16x32_f16(false, ah, false, bh, (short)0, c, false, false);
  c = __builtin_amdgcn_wmma_f32_16x16x32_f16(false, ah, false, bl, (short)0, c, false, false);
  c = __builtin_amdgcn_wmma_f32_16x16x32_f16(false, al, false, bh, (short)0, c, false, false);
  return c;
}

// One contiguous 32B LDS read -> whole 16-half fragment, no unpack VALU.
__device__ __forceinline__ v16h ld_v16h(const _Float16* p) {
  return *(const v16h*)p;
}

// A fragment from staging: elems 0-7 at p[0..7], elems 8-15 at p[16..23]
// (two 16B v8h reads landing in adjacent VGPR quads).
__device__ __forceinline__ v16h ld_a_frag(const _Float16* p) {
  union { v16h v; v8h c[2]; } u;
  u.c[0] = *(const v8h*)(p);
  u.c[1] = *(const v8h*)(p + 16);
  return u.v;
}

// Build an A fragment by splitting 16 fp32 values (unavoidable VALU: the
// source data is fp32 in global memory).
__device__ __forceinline__ void split_a_frag(const float* ta, v16h& ah, v16h& al) {
#pragma unroll
  for (int j = 0; j < 16; ++j) {
    _Float16 hi = (_Float16)ta[j];
    ah[j] = hi;
    al[j] = (_Float16)(ta[j] - (float)hi);
  }
}

// ---------------------------------------------------------------------------
// Kernel 0: d_out = node_feats  (residual baked into the accumulator)
// ---------------------------------------------------------------------------
__global__ void gnn_init_kernel(const float* __restrict__ nf, float* __restrict__ out, long n4) {
  long i = (long)blockIdx.x * blockDim.x + threadIdx.x;
  if (i < n4) ((float4*)out)[i] = ((const float4*)nf)[i];
}

// ---------------------------------------------------------------------------
// Kernel 1: edge MLP + atomic scatter-sum into d_out.
// One wave per 16-edge tile; 8 waves/block share the LDS weight planes.
// ---------------------------------------------------------------------------
__global__ void __launch_bounds__(256, 1)
gnn_edge_kernel(const float* __restrict__ node_feats,
                const float* __restrict__ edge_feats,
                const int*   __restrict__ src,
                const int*   __restrict__ dst,
                const float* __restrict__ W1a, const float* __restrict__ b1a,
                const float* __restrict__ W1b, const float* __restrict__ b1b,
                float* __restrict__ agg, int E) {
  extern __shared__ float4 smem_raw[];
  _Float16* sm = (_Float16*)smem_raw;

  const int tid = threadIdx.x;
  // Stage weights transposed (K-major per output column), split into hi/lo planes.
  for (int idx = tid; idx < 256 * 128; idx += 256) {
    const int k = idx >> 7, n = idx & 127;
    const float    w  = W1a[idx];
    const _Float16 hi = (_Float16)w;
    sm[kE_W1aHi + n * kStA + k] = hi;
    sm[kE_W1aLo + n * kStA + k] = (_Float16)(w - (float)hi);
  }
  for (int idx = tid; idx < 128 * 128; idx += 256) {
    const int k = idx >> 7, n = idx & 127;
    const float    w  = W1b[idx];
    const _Float16 hi = (_Float16)w;
    sm[kE_W1bHi + n * kStB + k] = hi;
    sm[kE_W1bLo + n * kStB + k] = (_Float16)(w - (float)hi);
  }
  __syncthreads();

  const int  wave = tid >> 5;
  const int  lane = tid & 31;
  const int  mrow = lane & 15;
  const int  hlf  = lane >> 4;  // lane half selects the K sub-chunks
  const long base = ((long)blockIdx.x * 8 + wave) * 16;
  if (base >= E) return;

  _Float16* t_hi = sm + kE_THi + wave * 16 * kStB;
  _Float16* t_lo = sm + kE_TLo + wave * 16 * kStB;

  const long   erow = base + mrow;
  const int    s    = src[erow];
  const float* an   = node_feats + (long)s * kInFeat;   // cat() first half (gather)
  const float* ae   = edge_feats + erow * kInFeat;      // cat() second half
  __builtin_prefetch(ae, 0, 0);                         // global_prefetch_b8

  const v8f vzero = {0.f, 0.f, 0.f, 0.f, 0.f, 0.f, 0.f, 0.f};
  v8f acc[8];
#pragma unroll
  for (int c = 0; c < 8; ++c) acc[c] = vzero;

  // -------- layer 1: [16,256] @ [256,128], K-loop of 8 x 32 --------
  for (int kk = 0; kk < 8; ++kk) {
    const int    k0 = kk * 32;
    const float* rp = (k0 < kInFeat) ? (an + k0) : (ae + (k0 - kInFeat));
    const int    c0 = hlf * 8;
    float ta[16];
    const float4 f0 = *(const float4*)(rp + c0);
    const float4 f1 = *(const float4*)(rp + c0 + 4);
    const float4 f2 = *(const float4*)(rp + c0 + 16);
    const float4 f3 = *(const float4*)(rp + c0 + 20);
    ta[0] = f0.x;  ta[1] = f0.y;  ta[2]  = f0.z;  ta[3]  = f0.w;
    ta[4] = f1.x;  ta[5] = f1.y;  ta[6]  = f1.z;  ta[7]  = f1.w;
    ta[8] = f2.x;  ta[9] = f2.y;  ta[10] = f2.z;  ta[11] = f2.w;
    ta[12] = f3.x; ta[13] = f3.y; ta[14] = f3.z;  ta[15] = f3.w;
    v16h ah, al;
    split_a_frag(ta, ah, al);
    const int kb = k0 + hlf * 8;
#pragma unroll
    for (int c = 0; c < 8; ++c) {
      const int row = (c * 16 + mrow) * kStA + kb;
      const v16h bh = ld_a_frag(sm + kE_W1aHi + row);
      const v16h bl = ld_a_frag(sm + kE_W1aLo + row);
      acc[c] = wmma3(ah, al, bh, bl, acc[c]);
    }
  }

  // bias + ReLU, stage split for layer 2 (C elem r -> row r + 8*hlf, col lane&15)
#pragma unroll
  for (int c = 0; c < 8; ++c) {
    const int   n    = c * 16 + mrow;
    const float bias = b1a[n];
#pragma unroll
    for (int r = 0; r < 8; ++r) {
      float v = acc[c][r] + bias;
      v = v > 0.f ? v : 0.f;
      const _Float16 hi = (_Float16)v;
      t_hi[(r + hlf * 8) * kStB + n] = hi;
      t_lo[(r + hlf * 8) * kStB + n] = (_Float16)(v - (float)hi);
    }
  }

  // -------- layer 2: [16,128] @ [128,128], K-loop of 4 x 32 --------
  v8f acc2[8];
#pragma unroll
  for (int c = 0; c < 8; ++c) acc2[c] = vzero;
  for (int kk = 0; kk < 4; ++kk) {
    const int  k0 = kk * 32;
    const int  ao = mrow * kStB + k0 + hlf * 8;
    const v16h ah = ld_a_frag(t_hi + ao);
    const v16h al = ld_a_frag(t_lo + ao);
    const int  kb = k0 + hlf * 8;
#pragma unroll
    for (int c = 0; c < 8; ++c) {
      const int row = (c * 16 + mrow) * kStB + kb;
      const v16h bh = ld_a_frag(sm + kE_W1bHi + row);
      const v16h bl = ld_a_frag(sm + kE_W1bLo + row);
      acc2[c] = wmma3(ah, al, bh, bl, acc2[c]);
    }
  }

  // bias + scatter-sum (f32 global atomic add, no return -> STOREcnt path)
  int dv[8];
#pragma unroll
  for (int r = 0; r < 8; ++r) dv[r] = dst[base + r + hlf * 8];
#pragma unroll
  for (int c = 0; c < 8; ++c) {
    const int   n    = c * 16 + mrow;
    const float bias = b1b[n];
#pragma unroll
    for (int r = 0; r < 8; ++r)
      unsafeAtomicAdd(agg + (long)dv[r] * kInFeat + n, acc2[c][r] + bias);
  }
}

// ---------------------------------------------------------------------------
// Kernel 2: node MLP, in place over d_out (h = agg + residual already there).
// ---------------------------------------------------------------------------
__global__ void __launch_bounds__(256, 1)
gnn_node_kernel(const float* __restrict__ W2a, const float* __restrict__ b2a,
                const float* __restrict__ W2b, const float* __restrict__ b2b,
                float* __restrict__ io, int N) {
  extern __shared__ float4 smem_raw[];
  _Float16* sm = (_Float16*)smem_raw;

  const int tid = threadIdx.x;
  for (int idx = tid; idx < 128 * 128; idx += 256) {
    const int k = idx >> 7, n = idx & 127;
    const float    wa  = W2a[idx];
    const _Float16 ha  = (_Float16)wa;
    sm[kN_W2aHi + n * kStB + k] = ha;
    sm[kN_W2aLo + n * kStB + k] = (_Float16)(wa - (float)ha);
    const float    wb  = W2b[idx];
    const _Float16 hb  = (_Float16)wb;
    sm[kN_W2bHi + n * kStB + k] = hb;
    sm[kN_W2bLo + n * kStB + k] = (_Float16)(wb - (float)hb);
  }
  __syncthreads();

  const int  wave = tid >> 5;
  const int  lane = tid & 31;
  const int  mrow = lane & 15;
  const int  hlf  = lane >> 4;
  const long base = ((long)blockIdx.x * 8 + wave) * 16;
  if (base >= N) return;

  _Float16* t_hi = sm + kN_THi + wave * 16 * kStB;
  _Float16* t_lo = sm + kN_TLo + wave * 16 * kStB;

  const float* hrow = io + (base + mrow) * (long)kInFeat;

  const v8f vzero = {0.f, 0.f, 0.f, 0.f, 0.f, 0.f, 0.f, 0.f};
  v8f acc[8];
#pragma unroll
  for (int c = 0; c < 8; ++c) acc[c] = vzero;

  // -------- layer A: relu(h @ W2a + b2a), K = 128 --------
  for (int kk = 0; kk < 4; ++kk) {
    const int    k0 = kk * 32;
    const float* rp = hrow + k0;
    const int    c0 = hlf * 8;
    float ta[16];
    const float4 f0 = *(const float4*)(rp + c0);
    const float4 f1 = *(const float4*)(rp + c0 + 4);
    const float4 f2 = *(const float4*)(rp + c0 + 16);
    const float4 f3 = *(const float4*)(rp + c0 + 20);
    ta[0] = f0.x;  ta[1] = f0.y;  ta[2]  = f0.z;  ta[3]  = f0.w;
    ta[4] = f1.x;  ta[5] = f1.y;  ta[6]  = f1.z;  ta[7]  = f1.w;
    ta[8] = f2.x;  ta[9] = f2.y;  ta[10] = f2.z;  ta[11] = f2.w;
    ta[12] = f3.x; ta[13] = f3.y; ta[14] = f3.z;  ta[15] = f3.w;
    v16h ah, al;
    split_a_frag(ta, ah, al);
    const int kb = k0 + hlf * 8;
#pragma unroll
    for (int c = 0; c < 8; ++c) {
      const int row = (c * 16 + mrow) * kStB + kb;
      const v16h bh = ld_a_frag(sm + kN_W2aHi + row);
      const v16h bl = ld_a_frag(sm + kN_W2aLo + row);
      acc[c] = wmma3(ah, al, bh, bl, acc[c]);
    }
  }

#pragma unroll
  for (int c = 0; c < 8; ++c) {
    const int   n    = c * 16 + mrow;
    const float bias = b2a[n];
#pragma unroll
    for (int r = 0; r < 8; ++r) {
      float v = acc[c][r] + bias;
      v = v > 0.f ? v : 0.f;
      const _Float16 hi = (_Float16)v;
      t_hi[(r + hlf * 8) * kStB + n] = hi;
      t_lo[(r + hlf * 8) * kStB + n] = (_Float16)(v - (float)hi);
    }
  }

  // -------- layer B: t @ W2b + b2b, K = 128, write back in place --------
  v8f acc2[8];
#pragma unroll
  for (int c = 0; c < 8; ++c) acc2[c] = vzero;
  for (int kk = 0; kk < 4; ++kk) {
    const int  k0 = kk * 32;
    const int  ao = mrow * kStB + k0 + hlf * 8;
    const v16h ah = ld_a_frag(t_hi + ao);
    const v16h al = ld_a_frag(t_lo + ao);
    const int  kb = k0 + hlf * 8;
#pragma unroll
    for (int c = 0; c < 8; ++c) {
      const int row = (c * 16 + mrow) * kStB + kb;
      const v16h bh = ld_a_frag(sm + kN_W2bHi + row);
      const v16h bl = ld_a_frag(sm + kN_W2bLo + row);
      acc2[c] = wmma3(ah, al, bh, bl, acc2[c]);
    }
  }

#pragma unroll
  for (int c = 0; c < 8; ++c) {
    const int   n    = c * 16 + mrow;
    const float bias = b2b[n];
#pragma unroll
    for (int r = 0; r < 8; ++r)
      io[(base + r + hlf * 8) * (long)kInFeat + n] = acc2[c][r] + bias;
  }
}

// ---------------------------------------------------------------------------
extern "C" void kernel_launch(void* const* d_in, const int* in_sizes, int n_in,
                              void* d_out, int out_size, void* d_ws, size_t ws_size,
                              hipStream_t stream) {
  const float* node_feats = (const float*)d_in[0];
  const float* edge_feats = (const float*)d_in[1];
  const int*   src        = (const int*)d_in[2];
  const int*   dst        = (const int*)d_in[3];
  const float* W1a = (const float*)d_in[4];
  const float* b1a = (const float*)d_in[5];
  const float* W1b = (const float*)d_in[6];
  const float* b1b = (const float*)d_in[7];
  const float* W2a = (const float*)d_in[8];
  const float* b2a = (const float*)d_in[9];
  const float* W2b = (const float*)d_in[10];
  const float* b2b = (const float*)d_in[11];
  float* out = (float*)d_out;

  const int E = in_sizes[2];             // 800000
  const int N = in_sizes[0] / kInFeat;   // 50000

  // Allow >64KB dynamic LDS (idempotent; runs immediately even under capture).
  (void)hipFuncSetAttribute(reinterpret_cast<const void*>(gnn_edge_kernel),
                            hipFuncAttributeMaxDynamicSharedMemorySize, kEdgeLds);
  (void)hipFuncSetAttribute(reinterpret_cast<const void*>(gnn_node_kernel),
                            hipFuncAttributeMaxDynamicSharedMemorySize, kNodeLds);

  const long n4 = (long)N * kInFeat / 4;
  gnn_init_kernel<<<(int)((n4 + 255) / 256), 256, 0, stream>>>(node_feats, out, n4);

  const int edgeTiles  = E / 16;
  const int edgeBlocks = (edgeTiles + 7) / 8;
  gnn_edge_kernel<<<edgeBlocks, 256, kEdgeLds, stream>>>(
      node_feats, edge_feats, src, dst, W1a, b1a, W1b, b1b, out, E);

  const int nodeTiles  = (N + 15) / 16;
  const int nodeBlocks = (nodeTiles + 7) / 8;
  gnn_node_kernel<<<nodeBlocks, 256, kNodeLds, stream>>>(W2a, b2a, W2b, b2b, out, N);
}
